// MyRNN_58463094833754
// MI455X (gfx1250) — compile-verified, matrix operand order checked by reading, not run on hardware
//
#include <hip/hip_runtime.h>

typedef __attribute__((ext_vector_type(16))) _Float16 v16h;
typedef __attribute__((ext_vector_type(8)))  _Float16 v8h;
typedef __attribute__((ext_vector_type(8)))  float    v8f;
typedef __attribute__((ext_vector_type(4)))  float    v4f;
typedef __attribute__((ext_vector_type(4)))  int      v4i;

#define SEQ   512
#define BATCH 128
#define INDIM 512
#define HID   1024
#define KTOT  (INDIM + HID)     // 1536: concatenated [x_t | h_{t-1}] reduction dim

#define BM 64                   // block M tile (batch rows)
#define BN 128                  // block N tile (hidden cols)
#define BK 64                   // K slice per staged chunk = 2 WMMA K-halves
#define XCH (INDIM / BK)        // 8 chunks sourced from x_t (fp32, needs convert)
#define NCH (KTOT / BK)         // 24 total chunks
#define LDS_STRIDE (BK + 8)     // 72 halves = 144B row: 16B-aligned, bank-conflict-free

// ---- CDNA5 async global->LDS path (ASYNCcnt), signature probe-confirmed ----
#if __has_builtin(__builtin_amdgcn_global_load_async_to_lds_b128)
#define HAVE_ASYNC 1
#else
#define HAVE_ASYNC 0
#endif

typedef __attribute__((address_space(1))) v4i* gptr_b128;
typedef __attribute__((address_space(3))) v4i* lptr_b128;

__device__ __forceinline__ void async_copy16(const _Float16* gsrc, _Float16* ldst) {
#if HAVE_ASYNC
    __builtin_amdgcn_global_load_async_to_lds_b128(
        (gptr_b128)(gsrc), (lptr_b128)(ldst), 0, 0);
#else
    *(v8h*)ldst = *(const v8h*)gsrc;
#endif
}

#if HAVE_ASYNC
#if __has_builtin(__builtin_amdgcn_s_wait_asynccnt)
#define WAIT_ASYNC(n) __builtin_amdgcn_s_wait_asynccnt((unsigned short)(n))
#else
#define WAIT_ASYNC(n) asm volatile("s_wait_asynccnt %0" :: "n"(n))
#endif
#else
#define WAIT_ASYNC(n) ((void)0)
#endif

// ---- one-time prep: W^T = concat(weight_x, weight_h) transposed to [HID][KTOT] f16
__global__ void prep_weights(const float* __restrict__ wx, const float* __restrict__ wh,
                             _Float16* __restrict__ Wt) {
    int idx = blockIdx.x * blockDim.x + threadIdx.x;     // n * KTOT + k
    if (idx >= HID * KTOT) return;
    int n = idx / KTOT, k = idx - n * KTOT;
    float v = (k < INDIM) ? wx[k * HID + n] : wh[(k - INDIM) * HID + n];
    Wt[idx] = (_Float16)v;
}

__global__ void prep_state(const float* __restrict__ s0, _Float16* __restrict__ h0) {
    int idx = blockIdx.x * blockDim.x + threadIdx.x;
    if (idx < BATCH * HID) h0[idx] = (_Float16)s0[idx];
}

// ---- one recurrence step: h_new = tanh([x_t | h] @ [Wx; Wh] + bias)
// GEMM M=128, N=1024, K=1536. Block = 8 waves; 64x128 tile; 2x2 WMMA tiles/wave;
// 24 double-buffered K chunks, 8 WMMAs per chunk, async global->LDS staging.
__global__ __launch_bounds__(256)
void rnn_step(const float*    __restrict__ x_t,    // [BATCH, INDIM] fp32
              const _Float16* __restrict__ Wt,     // [HID, KTOT]  f16 (B transposed)
              const _Float16* __restrict__ h_in,   // [BATCH, HID] f16
              const float*    __restrict__ bias,   // [HID] fp32
              float*          __restrict__ out_t,  // [BATCH, HID] fp32 (states[t])
              _Float16*       __restrict__ h_out)  // [BATCH, HID] f16 (next state)
{
    __shared__ __align__(16) _Float16 As[2][BM * LDS_STRIDE];   // 2 x 9 KB
    __shared__ __align__(16) _Float16 Bs[2][BN * LDS_STRIDE];   // 2 x 18 KB

    const int tid   = threadIdx.x;
    const int lane  = tid & 31;
    const int wave  = tid >> 5;
    const int row16 = lane & 15;                 // fragment row/col within 16
    const int half8 = (lane >> 4) ? 8 : 0;       // ISA lane->K (A/B) and lane->M (C/D) split

    const int wm = (wave & 1) * 2;               // M subtile base: 0 or 2   (of 4)
    const int wn = (wave >> 1) * 2;              // N subtile base: 0,2,4,6  (of 8)

    const int m_blk = blockIdx.y * BM;
    const int n_blk = blockIdx.x * BN;

    // staging: 16B chunks; A = 64x64 halves (2/thread), B = 128x64 halves (4/thread)
    const int a_row = tid >> 3;                  // 0..31  (+32 for q=1)
    const int a_col = (tid & 7) * 8;             // halves: 0..56

    v8f acc00 = {}, acc01 = {}, acc10 = {}, acc11 = {};

    auto stage_B = [&](int k0, int buf) {        // 4 asyncs
#pragma unroll
        for (int q = 0; q < 4; ++q) {
            int row = a_row + q * 32;
            async_copy16(Wt + (size_t)(n_blk + row) * KTOT + k0 + a_col,
                         &Bs[buf][row * LDS_STRIDE + a_col]);
        }
    };
    auto stage_Ah = [&](int k0, int buf) {       // 2 asyncs
#pragma unroll
        for (int q = 0; q < 2; ++q) {
            int row = a_row + q * 32;
            async_copy16(h_in + (size_t)(m_blk + row) * HID + (k0 - INDIM) + a_col,
                         &As[buf][row * LDS_STRIDE + a_col]);
        }
    };
    auto load_Ax = [&](int k0, v4f* f) {         // 4 x v4f, two rows
#pragma unroll
        for (int q = 0; q < 2; ++q) {
            const float* src = x_t + (size_t)(m_blk + a_row + q * 32) * INDIM + k0 + a_col;
            f[2 * q]     = *(const v4f*)src;
            f[2 * q + 1] = *(const v4f*)(src + 4);
        }
    };
    auto store_Ax = [&](int buf, const v4f* f) {
#pragma unroll
        for (int q = 0; q < 2; ++q) {
            v4f f0 = f[2 * q], f1 = f[2 * q + 1];
            v8h h;
            h[0] = (_Float16)f0[0]; h[1] = (_Float16)f0[1];
            h[2] = (_Float16)f0[2]; h[3] = (_Float16)f0[3];
            h[4] = (_Float16)f1[0]; h[5] = (_Float16)f1[1];
            h[6] = (_Float16)f1[2]; h[7] = (_Float16)f1[3];
            *(v8h*)&As[buf][(a_row + q * 32) * LDS_STRIDE + a_col] = h;
        }
    };
    auto consume = [&](int buf) {                // 8 WMMAs (2 K-halves x 2x2 tiles)
#pragma unroll
        for (int kh = 0; kh < 2; ++kh) {
            const int kc = kh * 32 + half8;
            union { v16h v; v8h h[2]; } a0, a1, b0, b1;
            const _Float16* ap0 = &As[buf][(wm * 16 + row16) * LDS_STRIDE + kc];
            const _Float16* ap1 = ap0 + 16 * LDS_STRIDE;
            a0.h[0] = *(const v8h*)ap0;        a0.h[1] = *(const v8h*)(ap0 + 16);
            a1.h[0] = *(const v8h*)ap1;        a1.h[1] = *(const v8h*)(ap1 + 16);
            const _Float16* bp0 = &Bs[buf][(wn * 16 + row16) * LDS_STRIDE + kc];
            const _Float16* bp1 = bp0 + 16 * LDS_STRIDE;
            b0.h[0] = *(const v8h*)bp0;        b0.h[1] = *(const v8h*)(bp0 + 16);
            b1.h[0] = *(const v8h*)bp1;        b1.h[1] = *(const v8h*)(bp1 + 16);
            acc00 = __builtin_amdgcn_wmma_f32_16x16x32_f16(false, a0.v, false, b0.v, (short)0, acc00, false, false);
            acc01 = __builtin_amdgcn_wmma_f32_16x16x32_f16(false, a0.v, false, b1.v, (short)0, acc01, false, false);
            acc10 = __builtin_amdgcn_wmma_f32_16x16x32_f16(false, a1.v, false, b0.v, (short)0, acc10, false, false);
            acc11 = __builtin_amdgcn_wmma_f32_16x16x32_f16(false, a1.v, false, b1.v, (short)0, acc11, false, false);
        }
    };

    v4f fx[4];

    // prologue: stage chunk 0 (x-type) into buf 0
    stage_B(0, 0);
    load_Ax(0, fx);
    store_Ax(0, fx);

    // region 1: consume x chunks 0..XCH-2, stage x chunk j+1 (4 asyncs + reg A path)
    for (int j = 0; j < XCH - 1; ++j) {
        const int k1 = (j + 1) * BK;
        load_Ax(k1, fx);                          // issue early: latency spans both barriers
        __syncthreads();                          // close prev iter's reads of buf (j+1)&1
        stage_B(k1, (j + 1) & 1);
        WAIT_ASYNC(4);                            // stage j's asyncs complete (in-order)
        __syncthreads();                          // buf j&1 fully visible
        consume(j & 1);
        store_Ax((j + 1) & 1, fx);                // convert + LDS store after WMMAs
    }
    // boundary: consume x chunk XCH-1, stage h chunk XCH (6 asyncs)
    {
        const int j = XCH - 1;
        __syncthreads();
        stage_B(XCH * BK, (j + 1) & 1);
        stage_Ah(XCH * BK, (j + 1) & 1);
        WAIT_ASYNC(6);
        __syncthreads();
        consume(j & 1);
    }
    // region 2: consume h chunks XCH..NCH-2, stage h chunk j+1 (6 asyncs)
    for (int j = XCH; j < NCH - 1; ++j) {
        const int k1 = (j + 1) * BK;
        __syncthreads();
        stage_B(k1, (j + 1) & 1);
        stage_Ah(k1, (j + 1) & 1);
        WAIT_ASYNC(6);
        __syncthreads();
        consume(j & 1);
    }
    // last chunk: nothing left to stage
    {
        __syncthreads();
        WAIT_ASYNC(0);
        __syncthreads();
        consume((NCH - 1) & 1);
    }

    // ---- epilogue: bias + tanh, write fp32 states and f16 next-state
    const int n0 = n_blk + wn * 16 + row16;
    const int n1 = n0 + 16;
    const float bb0 = bias[n0];
    const float bb1 = bias[n1];
#pragma unroll
    for (int v = 0; v < 8; ++v) {
        const int mA = m_blk + wm * 16 + half8 + v;    // tiles (wm, *)
        const int mB = mA + 16;                        // tiles (wm+1, *)
        float v00 = tanhf(acc00[v] + bb0);
        float v01 = tanhf(acc01[v] + bb1);
        float v10 = tanhf(acc10[v] + bb0);
        float v11 = tanhf(acc11[v] + bb1);
        out_t[(size_t)mA * HID + n0] = v00;  h_out[(size_t)mA * HID + n0] = (_Float16)v00;
        out_t[(size_t)mA * HID + n1] = v01;  h_out[(size_t)mA * HID + n1] = (_Float16)v01;
        out_t[(size_t)mB * HID + n0] = v10;  h_out[(size_t)mB * HID + n0] = (_Float16)v10;
        out_t[(size_t)mB * HID + n1] = v11;  h_out[(size_t)mB * HID + n1] = (_Float16)v11;
    }
}

extern "C" void kernel_launch(void* const* d_in, const int* in_sizes, int n_in,
                              void* d_out, int out_size, void* d_ws, size_t ws_size,
                              hipStream_t stream) {
    (void)in_sizes; (void)n_in; (void)out_size; (void)ws_size;
    const float* input = (const float*)d_in[0];   // [SEQ, BATCH, INDIM]
    const float* init  = (const float*)d_in[1];   // [BATCH, HID]
    const float* wx    = (const float*)d_in[2];   // [INDIM, HID]
    const float* wh    = (const float*)d_in[3];   // [HID, HID]
    const float* bias  = (const float*)d_in[4];   // [1, HID]
    float* out = (float*)d_out;                   // states [SEQ,BATCH,HID] ++ final [BATCH,HID]

    // workspace layout (~3.7 MB): Wt | h_buf0 | h_buf1
    char* ws = (char*)d_ws;
    _Float16* Wt = (_Float16*)ws;
    _Float16* h0 = (_Float16*)(ws + (size_t)HID * KTOT * sizeof(_Float16));
    _Float16* h1 = (_Float16*)(ws + (size_t)HID * KTOT * sizeof(_Float16)
                                  + (size_t)BATCH * HID * sizeof(_Float16));

    prep_weights<<<(HID * KTOT + 255) / 256, 256, 0, stream>>>(wx, wh, Wt);
    prep_state<<<(BATCH * HID + 255) / 256, 256, 0, stream>>>(init, h0);

    _Float16* hb[2] = { h0, h1 };
    dim3 grid(HID / BN, BATCH / BM);   // 8 x 2 blocks per step
    for (int t = 0; t < SEQ; ++t) {
        rnn_step<<<grid, 256, 0, stream>>>(
            input + (size_t)t * BATCH * INDIM, Wt, hb[t & 1], bias,
            out + (size_t)t * BATCH * HID, hb[(t + 1) & 1]);
    }
    // final_state == states[SEQ-1]
    (void)hipMemcpyAsync(out + (size_t)SEQ * BATCH * HID,
                         out + (size_t)(SEQ - 1) * BATCH * HID,
                         (size_t)BATCH * HID * sizeof(float),
                         hipMemcpyDeviceToDevice, stream);
}
